// MultiHeadAttention_42374147342779
// MI455X (gfx1250) — matrix operand order repaired and checked by reading, not compiled, not run on hardware
//
#include <hip/hip_runtime.h>
#include <hip/hip_bf16.h>

typedef __bf16 bf16;
typedef __attribute__((ext_vector_type(16))) __bf16 v16bf;
typedef __attribute__((ext_vector_type(8)))  __bf16 bf16x8;
typedef __attribute__((ext_vector_type(4)))  __bf16 bf16x4;
typedef __attribute__((ext_vector_type(8)))  float   v8f;
typedef __attribute__((ext_vector_type(4)))  unsigned u32x4;
typedef __attribute__((ext_vector_type(8)))  unsigned u32x8;

#define BATCH 4
#define SEQ   2048
#define DM    1024
#define NH    16
#define DEPTH 64
#define NEGV  (-1e9f)

// ---------------------------------------------------------------------------
// WMMA helper: D = A(16x32 bf16) * B(32x16 bf16) + C(16x16 f32)
// ---------------------------------------------------------------------------
static __device__ __forceinline__ v8f wmma_bf16(v16bf a, v16bf b, v8f c) {
  return __builtin_amdgcn_wmma_f32_16x16x32_bf16(
      false, a, false, b, (short)0, c, false, false);
}

// A fragment (16x32, 16-bit): lane = hi*16 + m holds row m.
// element i -> K = (i>>3)*16 + hi*8 + (i&7)  => two contiguous 16B runs.
static __device__ __forceinline__ v16bf frag_a(const bf16* tile, int rstride,
                                               int m, int k0, int hi) {
  const bf16* r = tile + (size_t)m * rstride + k0;
  bf16x8 lo = *(const bf16x8*)(r + hi * 8);
  bf16x8 hh = *(const bf16x8*)(r + 16 + hi * 8);
  v16bf a;
#pragma unroll
  for (int i = 0; i < 8; ++i) { a[i] = lo[i]; a[i + 8] = hh[i]; }
  return a;
}

// B fragment (32x16, 16-bit): lane holds column n = lane&15.
// element i -> K = (i>>3)*16 + hi*8 + (i&7).  rowk0 points at Bsrc[n-major row n][k0].
static __device__ __forceinline__ v16bf frag_b(const bf16* rowk0, int hi) {
  bf16x8 lo = *(const bf16x8*)(rowk0 + hi * 8);
  bf16x8 hh = *(const bf16x8*)(rowk0 + 16 + hi * 8);
  v16bf b;
#pragma unroll
  for (int i = 0; i < 8; ++i) { b[i] = lo[i]; b[i + 8] = hh[i]; }
  return b;
}

// ---------------------------------------------------------------------------
// TDM: 2D tile load Global -> LDS (D# per CDNA5 ISA ch.8; groups 0+1 only,
// VADDR2/3 = NULL).  data_size = 2 bytes.  Issued wave-uniform from SGPRs.
// ---------------------------------------------------------------------------
static __device__ __forceinline__ void tdm_load_2d(
    const void* gaddr, unsigned lds_off,
    unsigned tile_d0, unsigned tile_d1,
    unsigned tensor_d0, unsigned tensor_d1,
    unsigned long long stride0 /* elements */) {
  unsigned long long ga = (unsigned long long)(size_t)gaddr;
  u32x4 g0;
  g0[0] = 1u;                                            // count=1, user mode
  g0[1] = lds_off;                                       // lds_addr (bytes)
  g0[2] = (unsigned)ga;                                  // global_addr[31:0]
  g0[3] = (unsigned)((ga >> 32) & 0x01ffffffu) | (2u << 30); // addr[56:32]|type=2
  u32x8 g1;
  g1[0] = 1u << 16;                                      // data_size=1 -> 2B
  g1[1] = (tensor_d0 & 0xffffu) << 16;                   // tensor_dim0[15:0]
  g1[2] = ((tensor_d0 >> 16) & 0xffffu) | ((tensor_d1 & 0xffffu) << 16);
  g1[3] = ((tensor_d1 >> 16) & 0xffffu) | ((tile_d0 & 0xffffu) << 16);
  g1[4] = (tile_d1 & 0xffffu);                           // tile_dim1, tile_dim2=0
  g1[5] = (unsigned)stride0;                             // dim0_stride[31:0]
  g1[6] = (unsigned)((stride0 >> 32) & 0xffffu);         // dim0_stride[47:32]
  g1[7] = 0u;                                            // dim1_stride unused
  asm volatile("tensor_load_to_lds %0, %1" :: "s"(g0), "s"(g1) : "memory");
}

// ---------------------------------------------------------------------------
// f32 -> bf16 conversion (vectorized, grid-stride)
// ---------------------------------------------------------------------------
__global__ void cvt_f32_bf16_v4(const float* __restrict__ in,
                                bf16* __restrict__ out, int n4) {
  int i = blockIdx.x * blockDim.x + threadIdx.x;
  int stride = gridDim.x * blockDim.x;
  for (; i < n4; i += stride) {
    float4 f = ((const float4*)in)[i];
    bf16x4 o;
    o[0] = (bf16)f.x; o[1] = (bf16)f.y; o[2] = (bf16)f.z; o[3] = (bf16)f.w;
    ((bf16x4*)out)[i] = o;
  }
}

// ---------------------------------------------------------------------------
// QKV projection, 64x64 register-blocked: one wave computes 64 seq rows x
// full head width (64).  Per K-step: 4 A-frags + 4 B-frags -> 16 WMMAs.
// q,k stored [B,H,S,DEPTH]; v stored transposed [B,H,DEPTH,S].
// ---------------------------------------------------------------------------
__global__ void __launch_bounds__(128) qkv_kernel(
    const bf16* __restrict__ xb,
    const bf16* __restrict__ Wqb, const bf16* __restrict__ Wkb,
    const bf16* __restrict__ Wvb,
    const float* __restrict__ bq, const float* __restrict__ bk,
    const float* __restrict__ bv,
    bf16* __restrict__ q, bf16* __restrict__ k, bf16* __restrict__ vT) {
  const int lane = threadIdx.x & 31;
  const int ln = lane & 15;
  const int hi = lane >> 4;
  int wave = blockIdx.x * 4 + (threadIdx.x >> 5);
  const int mt = wave & 31;  wave >>= 5;   // 64-row seq tile
  const int h  = wave & 15;  wave >>= 4;
  const int b  = wave & 3;
  const int which = wave >> 2;             // 0=q 1=k 2=v
  if (which >= 3) return;

  const bf16*  W    = (which == 0) ? Wqb : (which == 1) ? Wkb : Wvb;
  const float* bias = (which == 0) ? bq  : (which == 1) ? bk  : bv;

  const bf16* xtile = xb + ((size_t)b * SEQ + (size_t)mt * 64) * DM;
  const bf16* wr[4];
#pragma unroll
  for (int c = 0; c < 4; ++c)
    wr[c] = W + ((size_t)h * DEPTH + c * 16 + ln) * DM;

  v8f acc[4][4];
#pragma unroll
  for (int j = 0; j < 4; ++j)
#pragma unroll
    for (int c = 0; c < 4; ++c) acc[j][c] = (v8f){};

  for (int k0 = 0; k0 < DM; k0 += 32) {
    v16bf a[4], bb[4];
#pragma unroll
    for (int j = 0; j < 4; ++j)
      a[j] = frag_a(xtile + (size_t)j * 16 * DM, DM, ln, k0, hi);
#pragma unroll
    for (int c = 0; c < 4; ++c) {
      __builtin_prefetch(wr[c] + k0 + 128, 0, 3);
      bb[c] = frag_b(wr[c] + k0, hi);
    }
#pragma unroll
    for (int j = 0; j < 4; ++j)
#pragma unroll
      for (int c = 0; c < 4; ++c) acc[j][c] = wmma_bf16(a[j], bb[c], acc[j][c]);
  }

  float bias_c[4];
#pragma unroll
  for (int c = 0; c < 4; ++c) bias_c[c] = bias[h * DEPTH + c * 16 + ln];

  if (which < 2) {
    bf16* dst = ((which == 0) ? q : k) +
                (((size_t)b * NH + h) * SEQ + (size_t)mt * 64) * DEPTH;
#pragma unroll
    for (int j = 0; j < 4; ++j)
#pragma unroll
      for (int c = 0; c < 4; ++c)
#pragma unroll
        for (int r = 0; r < 8; ++r)
          dst[(size_t)(j * 16 + 8 * hi + r) * DEPTH + c * 16 + ln] =
              (bf16)(acc[j][c][r] + bias_c[c]);
  } else {
    bf16* dstv = vT + (((size_t)b * NH + h) * DEPTH) * (size_t)SEQ;
#pragma unroll
    for (int c = 0; c < 4; ++c) {
      bf16* row = dstv + (size_t)(c * 16 + ln) * SEQ + (size_t)mt * 64;
#pragma unroll
      for (int j = 0; j < 4; ++j)
#pragma unroll
        for (int r = 0; r < 8; ++r)      // 8 consecutive bf16 -> b128 store
          row[j * 16 + 8 * hi + r] = (bf16)(acc[j][c][r] + bias_c[c]);
    }
  }
}

// ---------------------------------------------------------------------------
// Flash attention.  Workgroup = 4 waves = 4 query tiles of the same (b,h).
// Per 32-key chunk: wave0 TDM-loads K(32x64) -> LDS, wave1 TDM-loads the
// transposed-V chunk (64x32) -> LDS (s_wait_tensorcnt + barrier), then each
// wave: 4 score WMMAs from LDS, online softmax (shfl_xor row reductions),
// P restriped C-layout -> A-layout through private LDS, 4 accumulate WMMAs.
// ---------------------------------------------------------------------------
__global__ void __launch_bounds__(128) attn_kernel(
    const bf16* __restrict__ q, const bf16* __restrict__ k,
    const bf16* __restrict__ vT, const int* __restrict__ amask,
    bf16* __restrict__ mh) {
  __shared__ __align__(16) bf16 kTile[32][64];     // keys x depth   (4 KB)
  __shared__ __align__(16) bf16 vTile[64][32];     // depth x keys   (4 KB)
  __shared__ __align__(16) bf16 pbuf[4][16][48];   // per-wave P restripe

  const int wid  = threadIdx.x >> 5;
  const int lane = threadIdx.x & 31;
  const int ln = lane & 15;
  const int hi = lane >> 4;
  int blk = blockIdx.x;
  const int stq = blk & 31;  blk >>= 5;    // group of 4 query tiles
  const int h   = blk & 15;
  const int b   = blk >> 4;
  const int st  = stq * 4 + wid;           // this wave's 16-query tile

  const size_t bh = (size_t)b * NH + h;
  const bf16* qb = q  + bh * SEQ * DEPTH;
  const bf16* kb = k  + bh * SEQ * DEPTH;
  const bf16* vb = vT + bh * DEPTH * SEQ;
  const int*  mb = amask + (size_t)b * SEQ;

  const unsigned kOff = (unsigned)(size_t)(&kTile[0][0]);
  const unsigned vOff = (unsigned)(size_t)(&vTile[0][0]);

  // Q tile (16 x 64) as two A fragments, loaded once
  const bf16* qtile = qb + (size_t)st * 16 * DEPTH;
  v16bf qa0 = frag_a(qtile, DEPTH, ln, 0, hi);
  v16bf qa1 = frag_a(qtile, DEPTH, ln, 32, hi);

  float mrow[8], lrow[8];
  v8f acc[4];
#pragma unroll
  for (int r = 0; r < 8; ++r) { mrow[r] = -__builtin_inff(); lrow[r] = 0.f; }
#pragma unroll
  for (int e = 0; e < 4; ++e) acc[e] = (v8f){};

  for (int kt = 0; kt < SEQ; kt += 32) {
    // --- async tensor loads of the shared K/V chunk ---
    if (wid == 0) {
      tdm_load_2d(kb + (size_t)kt * DEPTH, kOff, 64, 32, 64, SEQ, 64ull);
      __builtin_amdgcn_s_wait_tensorcnt(0);
    } else if (wid == 1) {
      tdm_load_2d(vb + kt, vOff, 32, 64, SEQ, DEPTH, (unsigned long long)SEQ);
      __builtin_amdgcn_s_wait_tensorcnt(0);
    }
    __syncthreads();

    // --- scores for 32 keys: two 16-col subtiles, K=64 via 2x32 ---
    const bf16* kr0 = &kTile[ln][0];
    const bf16* kr1 = &kTile[16 + ln][0];
    v8f s0 = {}, s1 = {};
    s0 = wmma_bf16(qa0, frag_b(kr0, hi), s0);
    s0 = wmma_bf16(qa1, frag_b(kr0 + 32, hi), s0);
    s1 = wmma_bf16(qa0, frag_b(kr1, hi), s1);
    s1 = wmma_bf16(qa1, frag_b(kr1 + 32, hi), s1);

    const bool msk0 = (mb[kt + ln] == 0);
    const bool msk1 = (mb[kt + 16 + ln] == 0);

    float al[8];
#pragma unroll
    for (int r = 0; r < 8; ++r) {
      float a0 = msk0 ? NEGV : s0[r] * 0.125f;   // 1/sqrt(DEPTH)
      float a1 = msk1 ? NEGV : s1[r] * 0.125f;
      float vm = fmaxf(a0, a1);
      vm = fmaxf(vm, __shfl_xor(vm, 1));
      vm = fmaxf(vm, __shfl_xor(vm, 2));
      vm = fmaxf(vm, __shfl_xor(vm, 4));
      vm = fmaxf(vm, __shfl_xor(vm, 8));
      float mnew  = fmaxf(mrow[r], vm);
      float alpha = __expf(mrow[r] - mnew);
      mrow[r] = mnew;
      float p0 = __expf(a0 - mnew);
      float p1 = __expf(a1 - mnew);
      float ps = p0 + p1;
      ps += __shfl_xor(ps, 1);
      ps += __shfl_xor(ps, 2);
      ps += __shfl_xor(ps, 4);
      ps += __shfl_xor(ps, 8);
      lrow[r] = lrow[r] * alpha + ps;
      al[r] = alpha;
      pbuf[wid][8 * hi + r][ln]      = (bf16)p0;
      pbuf[wid][8 * hi + r][16 + ln] = (bf16)p1;
    }
#pragma unroll
    for (int e = 0; e < 4; ++e)
#pragma unroll
      for (int r = 0; r < 8; ++r) acc[e][r] *= al[r];

    __syncthreads();   // cross-lane P writes visible for A-layout reload

    const bf16* prow = &pbuf[wid][ln][0];
    bf16x8 plo = *(const bf16x8*)(prow + hi * 8);
    bf16x8 phh = *(const bf16x8*)(prow + 16 + hi * 8);
    v16bf pa;
#pragma unroll
    for (int i = 0; i < 8; ++i) { pa[i] = plo[i]; pa[i + 8] = phh[i]; }

#pragma unroll
    for (int e = 0; e < 4; ++e)
      acc[e] = wmma_bf16(pa, frag_b(&vTile[e * 16 + ln][0], hi), acc[e]);

    __syncthreads();   // protect kTile/vTile/pbuf before next chunk
  }

  bf16* out = mh + ((size_t)b * SEQ + (size_t)st * 16) * DM + h * DEPTH;
#pragma unroll
  for (int r = 0; r < 8; ++r) {
    float inv = 1.0f / lrow[r];
#pragma unroll
    for (int e = 0; e < 4; ++e)
      out[(size_t)(8 * hi + r) * DM + e * 16 + ln] = (bf16)(acc[e][r] * inv);
  }
}

// ---------------------------------------------------------------------------
// Output projection, 64x64 register-blocked: out = mh @ Wo^T + bo (f32 out)
// ---------------------------------------------------------------------------
__global__ void __launch_bounds__(128) oproj_kernel(
    const bf16* __restrict__ mh, const bf16* __restrict__ Wob,
    const float* __restrict__ bo, float* __restrict__ out) {
  const int lane = threadIdx.x & 31;
  const int ln = lane & 15;
  const int hi = lane >> 4;
  int wave = blockIdx.x * 4 + (threadIdx.x >> 5);
  const int nt = wave & 15;        // 64-col tile
  const int mt = wave >> 4;        // 64-row tile
  if (mt >= (BATCH * SEQ) / 64) return;

  const bf16* atile = mh + (size_t)mt * 64 * DM;
  const bf16* wr[4];
#pragma unroll
  for (int c = 0; c < 4; ++c)
    wr[c] = Wob + (size_t)(nt * 64 + c * 16 + ln) * DM;

  v8f acc[4][4];
#pragma unroll
  for (int j = 0; j < 4; ++j)
#pragma unroll
    for (int c = 0; c < 4; ++c) acc[j][c] = (v8f){};

  for (int k0 = 0; k0 < DM; k0 += 32) {
    v16bf a[4], bb[4];
#pragma unroll
    for (int j = 0; j < 4; ++j)
      a[j] = frag_a(atile + (size_t)j * 16 * DM, DM, ln, k0, hi);
#pragma unroll
    for (int c = 0; c < 4; ++c) {
      __builtin_prefetch(wr[c] + k0 + 128, 0, 3);
      bb[c] = frag_b(wr[c] + k0, hi);
    }
#pragma unroll
    for (int j = 0; j < 4; ++j)
#pragma unroll
      for (int c = 0; c < 4; ++c) acc[j][c] = wmma_bf16(a[j], bb[c], acc[j][c]);
  }

#pragma unroll
  for (int c = 0; c < 4; ++c) {
    float bias = bo[nt * 64 + c * 16 + ln];
#pragma unroll
    for (int j = 0; j < 4; ++j)
#pragma unroll
      for (int r = 0; r < 8; ++r)
        out[(size_t)(mt * 64 + j * 16 + 8 * hi + r) * DM + nt * 64 + c * 16 + ln] =
            acc[j][c][r] + bias;
  }
}

// ---------------------------------------------------------------------------
extern "C" void kernel_launch(void* const* d_in, const int* in_sizes, int n_in,
                              void* d_out, int out_size, void* d_ws,
                              size_t ws_size, hipStream_t stream) {
  const float* x  = (const float*)d_in[0];
  const int*   am = (const int*)d_in[1];
  const float* Wq = (const float*)d_in[2];
  const float* bq = (const float*)d_in[3];
  const float* Wk = (const float*)d_in[4];
  const float* bk = (const float*)d_in[5];
  const float* Wv = (const float*)d_in[6];
  const float* bv = (const float*)d_in[7];
  const float* Wo = (const float*)d_in[8];
  const float* bo = (const float*)d_in[9];
  float* out = (float*)d_out;

  const size_t XE  = (size_t)BATCH * SEQ * DM;
  const size_t WE  = (size_t)NH * DEPTH * DM;
  const size_t WOE = (size_t)DM * DM;
  const size_t QE  = (size_t)BATCH * NH * SEQ * DEPTH;

  bf16* p   = (bf16*)d_ws;
  bf16* xb  = p; p += XE;
  bf16* wqb = p; p += WE;
  bf16* wkb = p; p += WE;
  bf16* wvb = p; p += WE;
  bf16* wob = p; p += WOE;
  bf16* qd  = p; p += QE;
  bf16* kd  = p; p += QE;
  bf16* vtd = p; p += QE;
  bf16* mhd = p; p += XE;

  cvt_f32_bf16_v4<<<2048, 256, 0, stream>>>(x, xb, (int)(XE / 4));
  cvt_f32_bf16_v4<<<512, 256, 0, stream>>>(Wq, wqb, (int)(WE / 4));
  cvt_f32_bf16_v4<<<512, 256, 0, stream>>>(Wk, wkb, (int)(WE / 4));
  cvt_f32_bf16_v4<<<512, 256, 0, stream>>>(Wv, wvb, (int)(WE / 4));
  cvt_f32_bf16_v4<<<512, 256, 0, stream>>>(Wo, wob, (int)(WOE / 4));

  // 3 matrices * 4 batch * 16 heads * 32 row-tiles = 6144 waves
  qkv_kernel<<<6144 / 4, 128, 0, stream>>>(xb, wqb, wkb, wvb, bq, bk, bv,
                                           qd, kd, vtd);
  // 4 batch * 16 heads * 32 tile-groups = 2048 workgroups
  attn_kernel<<<2048, 128, 0, stream>>>(qd, kd, vtd, am, mhd);
  // 128 row-tiles * 16 col-tiles = 2048 waves
  oproj_kernel<<<2048 / 4, 128, 0, stream>>>(mhd, wob, bo, out);
}
